// InteractionPPBlock_23149873725723
// MI455X (gfx1250) — matrix operand order repaired and checked by reading, not compiled, not run on hardware
//
#include <hip/hip_runtime.h>

#define E_N   200000
#define E_PAD 200064   // 1563 * 128
#define T_N   2000000
#define EMBD  128
#define INTD  64
#define NRBF_ 6
#define NSBF_ 42
#define BAS_  8

typedef __attribute__((ext_vector_type(16))) __bf16 bf16x16;
typedef __attribute__((ext_vector_type(8)))  float  f32x8;

union AFrag {
    bf16x16      v;
    unsigned int u[8];
    uint4        q[2];
};

__device__ inline f32x8 wmma_bf16(bf16x16 a, bf16x16 b, f32x8 c) {
    return __builtin_amdgcn_wmma_f32_16x16x32_bf16(
        /*neg_a=*/false, a, /*neg_b=*/false, b,
        /*c_mod=*/(short)0, c, /*reuse_a=*/false, /*reuse_b=*/false);
}

__device__ inline float silu_f(float v) {
    return v * __builtin_amdgcn_rcpf(1.0f + __expf(-v));
}

// Relaxed agent-scope f32 atomic add -> hardware global_atomic_add_f32.
__device__ inline void atomic_add_f32(float* p, float v) {
    __hip_atomic_fetch_add(p, v, __ATOMIC_RELAXED, __HIP_MEMORY_SCOPE_AGENT);
}

// A-matrix 16x32 bf16 fragment from an LDS tile (row-major, pitch 130 halves).
// ISA layout: lanes 0-15 row M=lane, K={0..7,16..23}; lanes 16-31 same row, K={8..15,24..31}.
__device__ inline bf16x16 load_a_frag(const __bf16* tile, int row0, int kt, int lane) {
    const int half = lane >> 4;
    const __bf16* p = tile + (row0 + (lane & 15)) * 130 + kt * 32 + half * 8;
    AFrag f;
    f.u[0] = *(const unsigned int*)(p + 0);
    f.u[1] = *(const unsigned int*)(p + 2);
    f.u[2] = *(const unsigned int*)(p + 4);
    f.u[3] = *(const unsigned int*)(p + 6);
    f.u[4] = *(const unsigned int*)(p + 16);
    f.u[5] = *(const unsigned int*)(p + 18);
    f.u[6] = *(const unsigned int*)(p + 20);
    f.u[7] = *(const unsigned int*)(p + 22);
    return f.v;
}

// B-matrix 32x16 bf16 fragment from the pre-swizzled weight buffer.
__device__ inline bf16x16 load_b_frag(const __bf16* fb, int KT, int nt, int kt, int lane) {
    const uint4* q = (const uint4*)(fb + (size_t)(((nt * KT + kt) * 32 + lane)) * 16);
    AFrag f;
    f.q[0] = q[0];
    f.q[1] = q[1];
    return f.v;
}

// Write one wave's 16x128 f32 result as bf16 into the LDS tile (rows wave-private).
__device__ inline void store_tile(__bf16* tile, int rB, int lane, const float* vals) {
    const int half = lane >> 4, col = lane & 15;
    __bf16* tp = tile + (rB + 8 * half) * 130 + col;
#pragma unroll
    for (int nt = 0; nt < 8; nt++)
#pragma unroll
        for (int i = 0; i < 8; i++)
            tp[i * 130 + nt * 16] = (__bf16)vals[nt * 8 + i];
}

// (16x128)@(128x128) GEMM + bias + SiLU; A-frags loaded to regs first
// so the caller may overwrite the tile afterwards.
__device__ inline void gemm128_silu(const __bf16* tile, int row0, int lane,
                                    const __bf16* __restrict__ Wf,
                                    const float* __restrict__ bias,
                                    float* outv) {
    bf16x16 a[4];
#pragma unroll
    for (int kt = 0; kt < 4; kt++) a[kt] = load_a_frag(tile, row0, kt, lane);
    const int col = lane & 15;
#pragma unroll
    for (int nt = 0; nt < 8; nt++) {
        float b = bias[nt * 16 + col];
        f32x8 acc;
#pragma unroll
        for (int i = 0; i < 8; i++) acc[i] = b;
#pragma unroll
        for (int kt = 0; kt < 4; kt++)
            acc = wmma_bf16(a[kt], load_b_frag(Wf, 4, nt, kt, lane), acc);
#pragma unroll
        for (int i = 0; i < 8; i++) outv[nt * 8 + i] = silu_f(acc[i]);
    }
}

// ---------------- prep kernels ----------------

__global__ __launch_bounds__(256) void k_fold_rbf(const float* __restrict__ W1,
                                                  const float* __restrict__ W2,
                                                  float* __restrict__ out) {
    int idx = blockIdx.x * 256 + threadIdx.x;
    if (idx < NRBF_ * EMBD) {
        int j = idx >> 7, n = idx & 127;
        float a = 0.f;
#pragma unroll
        for (int b = 0; b < BAS_; b++) a += W1[j * BAS_ + b] * W2[b * EMBD + n];
        out[idx] = a;
    }
}

// Reorder W[K][N] f32 into bf16 WMMA B-fragment layout.
__global__ __launch_bounds__(256) void k_conv_frag(const float* __restrict__ W,
                                                   __bf16* __restrict__ dst, int K, int N) {
    int total = K * N;
    for (int idx = blockIdx.x * 256 + threadIdx.x; idx < total; idx += gridDim.x * 256) {
        int k = idx / N, n = idx - k * N;
        int kt = k >> 5, rk = k & 31, nt = n >> 4, col = n & 15;
        int half = (rk >> 3) & 1, e = (rk & 7) | ((rk >> 4) << 3);
        int lane = half * 16 + col;
        int KT = K >> 5;
        dst[(((nt * KT + kt) * 32 + lane) << 4) + e] = (__bf16)W[idx];
    }
}

// ---------------- edge front: x_ji, x_kj*r, down-projection ----------------

__global__ __launch_bounds__(256) void k_edge_front(
    const float* __restrict__ x, const float* __restrict__ rbf,
    const float* __restrict__ Wrbf,
    const __bf16* __restrict__ Wji_f, const float* __restrict__ bji,
    const __bf16* __restrict__ Wkj_f, const float* __restrict__ bkj,
    const __bf16* __restrict__ Wdn_f,
    float* __restrict__ xji, float* __restrict__ dbuf, float* __restrict__ m) {
    __shared__ __bf16 tile[128 * 130];
    __shared__ float  srbf[128 * NRBF_];
    const int  tid = threadIdx.x;
    const long rowBase = (long)blockIdx.x * 128;
    const bool full = (rowBase + 128 <= (long)E_N);  // wave-uniform

    // zero the segment-sum accumulator (ws is padded -> unguarded)
    {
        float* mz = m + rowBase * INTD;
        for (int i = tid; i < 128 * 64; i += 256) mz[i] = 0.f;
    }
    // stage x -> bf16 LDS; stage rbf
    if (full) {
        const float* xb = x + rowBase * EMBD;
        for (int i = tid; i < 128 * 128; i += 256)
            tile[(i >> 7) * 130 + (i & 127)] = (__bf16)xb[i];
        const float* rb_ = rbf + rowBase * NRBF_;
        for (int i = tid; i < 128 * NRBF_; i += 256) srbf[i] = rb_[i];
    } else {
        for (int i = tid; i < 128 * 128; i += 256) {
            int r = i >> 7, c = i & 127;
            long e = rowBase + r;
            tile[r * 130 + c] = (__bf16)((e < E_N) ? x[e * EMBD + c] : 0.f);
        }
        for (int i = tid; i < 128 * NRBF_; i += 256) {
            int r = i / NRBF_, c = i - r * NRBF_;
            long e = rowBase + r;
            srbf[i] = (e < E_N) ? rbf[e * NRBF_ + c] : 0.f;
        }
    }
    __syncthreads();

    const int lane = tid & 31, wave = tid >> 5;
    const int rB = wave * 16, half = lane >> 4, col = lane & 15;
    const long laneRow = rowBase + rB + 8 * half;  // first of this lane-half's 8 rows

    bf16x16 ax[4];
#pragma unroll
    for (int kt = 0; kt < 4; kt++) ax[kt] = load_a_frag(tile, rB, kt, lane);

    float rb[8][NRBF_];
    {
        const float* sb = srbf + (rB + 8 * half) * NRBF_;
#pragma unroll
        for (int i = 0; i < 8; i++)
#pragma unroll
            for (int j = 0; j < NRBF_; j++) rb[i][j] = sb[i * NRBF_ + j];
    }

    // x_ji = silu(x @ W_ji + b_ji) -> global (padded ws, unguarded)
    {
        float* po = xji + laneRow * EMBD + col;
#pragma unroll
        for (int nt = 0; nt < 8; nt++) {
            float b = bji[nt * 16 + col];
            f32x8 acc;
#pragma unroll
            for (int i = 0; i < 8; i++) acc[i] = b;
#pragma unroll
            for (int kt = 0; kt < 4; kt++)
                acc = wmma_bf16(ax[kt], load_b_frag(Wji_f, 4, nt, kt, lane), acc);
#pragma unroll
            for (int i = 0; i < 8; i++) po[i * EMBD + nt * 16] = silu_f(acc[i]);
        }
    }

    // x_kj = silu(x @ W_kj + b_kj) * (rbf @ W_rbf) -> back into tile (own rows)
    {
        __bf16* tp = tile + (rB + 8 * half) * 130 + col;
#pragma unroll
        for (int nt = 0; nt < 8; nt++) {
            float b = bkj[nt * 16 + col];
            f32x8 acc;
#pragma unroll
            for (int i = 0; i < 8; i++) acc[i] = b;
#pragma unroll
            for (int kt = 0; kt < 4; kt++)
                acc = wmma_bf16(ax[kt], load_b_frag(Wkj_f, 4, nt, kt, lane), acc);
            float wr[NRBF_];
#pragma unroll
            for (int j = 0; j < NRBF_; j++) wr[j] = Wrbf[j * EMBD + nt * 16 + col];
#pragma unroll
            for (int i = 0; i < 8; i++) {
                float rv = 0.f;
#pragma unroll
                for (int j = 0; j < NRBF_; j++) rv += rb[i][j] * wr[j];
                tp[i * 130 + nt * 16] = (__bf16)(silu_f(acc[i]) * rv);
            }
        }
    }

    // d = silu(x_kj @ W_down) -> global (padded ws, unguarded)
    {
        bf16x16 ak[4];
#pragma unroll
        for (int kt = 0; kt < 4; kt++) ak[kt] = load_a_frag(tile, rB, kt, lane);
        float* pd = dbuf + laneRow * INTD + col;
#pragma unroll
        for (int nt = 0; nt < 4; nt++) {
            f32x8 acc;
#pragma unroll
            for (int i = 0; i < 8; i++) acc[i] = 0.f;
#pragma unroll
            for (int kt = 0; kt < 4; kt++)
                acc = wmma_bf16(ak[kt], load_b_frag(Wdn_f, 4, nt, kt, lane), acc);
#pragma unroll
            for (int i = 0; i < 8; i++) pd[i * INTD + nt * 16] = silu_f(acc[i]);
        }
    }
}

// ---------------- triplet side: gather * sbf-basis, scatter-add ----------------

__global__ __launch_bounds__(256) void k_triplet(
    const float* __restrict__ sbf, const int* __restrict__ id_exp,
    const int* __restrict__ id_red,
    const float* __restrict__ Ws1, const float* __restrict__ Ws2,
    const float* __restrict__ dbuf, float* __restrict__ m) {
    __shared__ float s1w[NSBF_ * BAS_];
    __shared__ float s2w[BAS_ * INTD];
    for (int i = threadIdx.x; i < NSBF_ * BAS_; i += 256) s1w[i] = Ws1[i];
    for (int i = threadIdx.x; i < BAS_ * INTD; i += 256) s2w[i] = Ws2[i];
    __syncthreads();

    long t = (long)blockIdx.x * 256 + threadIdx.x;
    if (t >= T_N) return;

    const float* sp = sbf + t * NSBF_;
    float sb[NSBF_];
#pragma unroll
    for (int i = 0; i < NSBF_ / 2; i++) {  // 8-byte aligned pair loads
        float2 v = ((const float2*)sp)[i];
        sb[2 * i] = v.x;
        sb[2 * i + 1] = v.y;
    }
    float s1[BAS_];
#pragma unroll
    for (int b = 0; b < BAS_; b++) {
        float a = 0.f;
#pragma unroll
        for (int j = 0; j < NSBF_; j++) a += sb[j] * s1w[j * BAS_ + b];
        s1[b] = a;
    }

    const int e  = id_exp[t];
    const int ji = id_red[t];
    const float* dp = dbuf + (long)e * 64;
    float* mp = m + (long)ji * 64;
#pragma unroll
    for (int j4 = 0; j4 < 16; j4++) {
        float4 dv = ((const float4*)dp)[j4];
#pragma unroll
        for (int c = 0; c < 4; c++) {
            int j = 4 * j4 + c;
            float s = 0.f;
#pragma unroll
            for (int b = 0; b < BAS_; b++) s += s1[b] * s2w[b * INTD + j];
            float val = ((c == 0) ? dv.x : (c == 1) ? dv.y : (c == 2) ? dv.z : dv.w) * s;
            atomic_add_f32(mp + j, val);
        }
    }
}

// ---------------- edge back: up-projection + residual stack ----------------

__global__ __launch_bounds__(256) void k_edge_back(
    const float* __restrict__ x, const float* __restrict__ xji,
    const float* __restrict__ m,
    const __bf16* __restrict__ Wup_f,
    const __bf16* __restrict__ Wb1_f, const float* __restrict__ bb1,
    const __bf16* __restrict__ Wb2_f, const float* __restrict__ bb2,
    const __bf16* __restrict__ Wfin_f, const float* __restrict__ bfin,
    const __bf16* __restrict__ Wa11_f, const float* __restrict__ ba11,
    const __bf16* __restrict__ Wa12_f, const float* __restrict__ ba12,
    const __bf16* __restrict__ Wa21_f, const float* __restrict__ ba21,
    const __bf16* __restrict__ Wa22_f, const float* __restrict__ ba22,
    float* __restrict__ out) {
    __shared__ __bf16 tile[128 * 130];
    const int  tid = threadIdx.x;
    const long rowBase = (long)blockIdx.x * 128;
    const bool full = (rowBase + 128 <= (long)E_N);  // wave-uniform

    // stage m (padded ws, unguarded)
    {
        const float* mb = m + rowBase * INTD;
        for (int i = tid; i < 128 * 64; i += 256)
            tile[(i >> 6) * 130 + (i & 63)] = (__bf16)mb[i];
    }
    __syncthreads();

    const int lane = tid & 31, wave = tid >> 5;
    const int rB = wave * 16, half = lane >> 4, col = lane & 15;
    const long laneRow = rowBase + rB + 8 * half;

    float h[64], tv[64];

    // x2 = x_ji + silu(m @ W_up)
    {
        bf16x16 am[2];
        am[0] = load_a_frag(tile, rB, 0, lane);
        am[1] = load_a_frag(tile, rB, 1, lane);
        const float* pxj = xji + laneRow * EMBD + col;
#pragma unroll
        for (int nt = 0; nt < 8; nt++) {
            f32x8 acc;
#pragma unroll
            for (int i = 0; i < 8; i++) acc[i] = 0.f;
#pragma unroll
            for (int kt = 0; kt < 2; kt++)
                acc = wmma_bf16(am[kt], load_b_frag(Wup_f, 2, nt, kt, lane), acc);
#pragma unroll
            for (int i = 0; i < 8; i++)
                h[nt * 8 + i] = pxj[i * EMBD + nt * 16] + silu_f(acc[i]);
        }
    }
    store_tile(tile, rB, lane, h);

    // residual before skip
    gemm128_silu(tile, rB, lane, Wb1_f, bb1, tv);
    store_tile(tile, rB, lane, tv);
    gemm128_silu(tile, rB, lane, Wb2_f, bb2, tv);
#pragma unroll
    for (int j = 0; j < 64; j++) h[j] += tv[j];
    store_tile(tile, rB, lane, h);

    // final before skip + skip connection (x is harness-owned: uniform guard)
    gemm128_silu(tile, rB, lane, Wfin_f, bfin, tv);
    if (full) {
        const float* px = x + laneRow * EMBD + col;
#pragma unroll
        for (int nt = 0; nt < 8; nt++)
#pragma unroll
            for (int i = 0; i < 8; i++)
                h[nt * 8 + i] = px[i * EMBD + nt * 16] + tv[nt * 8 + i];
    } else {
#pragma unroll
        for (int nt = 0; nt < 8; nt++)
#pragma unroll
            for (int i = 0; i < 8; i++) {
                long e = laneRow + i;
                float xv = (e < E_N) ? x[e * EMBD + nt * 16 + col] : 0.f;
                h[nt * 8 + i] = xv + tv[nt * 8 + i];
            }
    }
    store_tile(tile, rB, lane, h);

    // residual after skip 1
    gemm128_silu(tile, rB, lane, Wa11_f, ba11, tv);
    store_tile(tile, rB, lane, tv);
    gemm128_silu(tile, rB, lane, Wa12_f, ba12, tv);
#pragma unroll
    for (int j = 0; j < 64; j++) h[j] += tv[j];
    store_tile(tile, rB, lane, h);

    // residual after skip 2
    gemm128_silu(tile, rB, lane, Wa21_f, ba21, tv);
    store_tile(tile, rB, lane, tv);
    gemm128_silu(tile, rB, lane, Wa22_f, ba22, tv);
#pragma unroll
    for (int j = 0; j < 64; j++) h[j] += tv[j];

    // final store (harness-owned out: uniform guard)
    if (full) {
        float* po = out + laneRow * EMBD + col;
#pragma unroll
        for (int nt = 0; nt < 8; nt++)
#pragma unroll
            for (int i = 0; i < 8; i++) po[i * EMBD + nt * 16] = h[nt * 8 + i];
    } else {
#pragma unroll
        for (int nt = 0; nt < 8; nt++)
#pragma unroll
            for (int i = 0; i < 8; i++) {
                long e = laneRow + i;
                if (e < E_N) out[e * EMBD + nt * 16 + col] = h[nt * 8 + i];
            }
    }
}

// ---------------- launch ----------------

extern "C" void kernel_launch(void* const* d_in, const int* in_sizes, int n_in,
                              void* d_out, int out_size, void* d_ws, size_t ws_size,
                              hipStream_t stream) {
    const float* x     = (const float*)d_in[0];
    const float* rbf   = (const float*)d_in[1];
    const float* sbf   = (const float*)d_in[2];
    const int*   idk   = (const int*)d_in[3];
    const int*   idj   = (const int*)d_in[4];
    const float* Wr1   = (const float*)d_in[5];
    const float* Wr2   = (const float*)d_in[6];
    const float* Ws1   = (const float*)d_in[7];
    const float* Ws2   = (const float*)d_in[8];
    const float* W_ji  = (const float*)d_in[9];
    const float* b_ji  = (const float*)d_in[10];
    const float* W_kj  = (const float*)d_in[11];
    const float* b_kj  = (const float*)d_in[12];
    const float* W_dn  = (const float*)d_in[13];
    const float* W_up  = (const float*)d_in[14];
    const float* Wb1   = (const float*)d_in[15];
    const float* bb1   = (const float*)d_in[16];
    const float* Wb2   = (const float*)d_in[17];
    const float* bb2   = (const float*)d_in[18];
    const float* Wfin  = (const float*)d_in[19];
    const float* bfin  = (const float*)d_in[20];
    const float* Wa11  = (const float*)d_in[21];
    const float* ba11  = (const float*)d_in[22];
    const float* Wa12  = (const float*)d_in[23];
    const float* ba12  = (const float*)d_in[24];
    const float* Wa21  = (const float*)d_in[25];
    const float* ba21  = (const float*)d_in[26];
    const float* Wa22  = (const float*)d_in[27];
    const float* ba22  = (const float*)d_in[28];
    float* out = (float*)d_out;

    // workspace layout (row-padded to E_PAD so stores need no guards)
    float* p_xji  = (float*)d_ws;                      // E_PAD*128 f32
    float* p_d    = p_xji + (size_t)E_PAD * EMBD;      // E_PAD*64 f32
    float* p_m    = p_d + (size_t)E_PAD * INTD;        // E_PAD*64 f32
    float* p_wrbf = p_m + (size_t)E_PAD * INTD;        // 6*128 f32
    __bf16* fr = (__bf16*)(p_wrbf + NRBF_ * EMBD);
    __bf16* Wji_f  = fr;             fr += 16384;
    __bf16* Wkj_f  = fr;             fr += 16384;
    __bf16* Wdn_f  = fr;             fr += 8192;
    __bf16* Wup_f  = fr;             fr += 8192;
    __bf16* Wb1_f  = fr;             fr += 16384;
    __bf16* Wb2_f  = fr;             fr += 16384;
    __bf16* Wfin_f = fr;             fr += 16384;
    __bf16* Wa11_f = fr;             fr += 16384;
    __bf16* Wa12_f = fr;             fr += 16384;
    __bf16* Wa21_f = fr;             fr += 16384;
    __bf16* Wa22_f = fr;             fr += 16384;

    k_fold_rbf<<<3, 256, 0, stream>>>(Wr1, Wr2, p_wrbf);
    k_conv_frag<<<64, 256, 0, stream>>>(W_ji, Wji_f, 128, 128);
    k_conv_frag<<<64, 256, 0, stream>>>(W_kj, Wkj_f, 128, 128);
    k_conv_frag<<<32, 256, 0, stream>>>(W_dn, Wdn_f, 128, 64);
    k_conv_frag<<<32, 256, 0, stream>>>(W_up, Wup_f, 64, 128);
    k_conv_frag<<<64, 256, 0, stream>>>(Wb1, Wb1_f, 128, 128);
    k_conv_frag<<<64, 256, 0, stream>>>(Wb2, Wb2_f, 128, 128);
    k_conv_frag<<<64, 256, 0, stream>>>(Wfin, Wfin_f, 128, 128);
    k_conv_frag<<<64, 256, 0, stream>>>(Wa11, Wa11_f, 128, 128);
    k_conv_frag<<<64, 256, 0, stream>>>(Wa12, Wa12_f, 128, 128);
    k_conv_frag<<<64, 256, 0, stream>>>(Wa21, Wa21_f, 128, 128);
    k_conv_frag<<<64, 256, 0, stream>>>(Wa22, Wa22_f, 128, 128);

    const int EB = E_PAD / 128;         // 1563
    const int TB = (T_N + 255) / 256;   // 7813
    k_edge_front<<<EB, 256, 0, stream>>>(x, rbf, p_wrbf, Wji_f, b_ji, Wkj_f, b_kj,
                                         Wdn_f, p_xji, p_d, p_m);
    k_triplet<<<TB, 256, 0, stream>>>(sbf, idk, idj, Ws1, Ws2, p_d, p_m);
    k_edge_back<<<EB, 256, 0, stream>>>(x, p_xji, p_m, Wup_f,
                                        Wb1_f, bb1, Wb2_f, bb2, Wfin_f, bfin,
                                        Wa11_f, ba11, Wa12_f, ba12,
                                        Wa21_f, ba21, Wa22_f, ba22, out);
    (void)in_sizes; (void)n_in; (void)out_size; (void)ws_size;
}
